// _ChannelLSTMT_3693671875179
// MI455X (gfx1250) — matrix operand order repaired and checked by reading, not compiled, not run on hardware
//
#include <hip/hip_runtime.h>

typedef unsigned short u16;
typedef __attribute__((ext_vector_type(4)))  unsigned int v4u;
typedef __attribute__((ext_vector_type(16))) __bf16      v16bf;
typedef __attribute__((ext_vector_type(8)))  float       v8f;

#define GAS __attribute__((address_space(1)))   // global address space

#define B_    4
#define C_    64
#define F_    257
#define T_    256
#define H_    256
#define OUT_  64
#define N_    (B_ * F_)   // 1028 sequences
#define NPAD_ 1040        // 65 tiles of 16
#define NBLK_ 65

__device__ __forceinline__ u16 f2bf(float f) {
  unsigned int u = __builtin_bit_cast(unsigned int, f);
  u += 0x7FFFu + ((u >> 16) & 1u);   // round-to-nearest-even
  return (u16)(u >> 16);
}

// Optimization barrier: launder a uniform pointer through an SGPR so loads based on
// it cannot be proven loop-invariant (prevents LICM hoisting -> register spills).
// Returns an explicit addrspace(1) pointer so loads lower to global_load (not flat),
// keeping the scalar-base + voffset form and pure LOADcnt tracking.
__device__ __forceinline__ const GAS u16* launder_g(const u16* p) {
  unsigned long long v = (unsigned long long)p;
  asm volatile("" : "+s"(v));
  return (const GAS u16*)v;
}

union FragU { v4u q[2]; v16bf bf; };

// A-matrix 16x32 bf16 fragment (ISA 7.12.2): lane<16 holds K{k0..k0+7, k0+16..k0+23},
// lane>=16 holds K{k0+8..k0+15, k0+24..k0+31}. Caller passes p = row + k0 + hi*8.
__device__ __forceinline__ v16bf load_frag_a(const u16* p) {
  FragU f;
  f.q[0] = *(const v4u*)(p);
  f.q[1] = *(const v4u*)(p + 16);
  return f.bf;
}
// B-matrix 32x16 bf16 fragment: lane<16 = col r K{k0..k0+7, k0+16..k0+23}, lane>=16 = col r
// K{k0+8..k0+15, k0+24..k0+31}. Caller passes p = col_row_ptr + k0 + hi*8.
__device__ __forceinline__ v16bf load_frag_b_g(const GAS u16* p) {
  FragU f;
  f.q[0] = *(const GAS v4u*)(p);
  f.q[1] = *(const GAS v4u*)(p + 16);
  return f.bf;
}

__device__ __forceinline__ v8f wmma_bf16(v16bf a, v16bf b, v8f c) {
  return __builtin_amdgcn_wmma_f32_16x16x32_bf16(false, a, false, b, (short)0, c,
                                                 false, false);
}

// Fast activations: v_exp_f32 + v_rcp_f32 (TRANS ops co-execute with WMMA)
__device__ __forceinline__ float sigm(float x) {
  return __builtin_amdgcn_rcpf(1.f + __expf(-x));
}
__device__ __forceinline__ float tanh_fast(float x) {
  x = fminf(fmaxf(x, -15.f), 15.f);
  float e = __expf(2.f * x);
  return (e - 1.f) * __builtin_amdgcn_rcpf(e + 1.f);
}

// ---------------- Kernel 1: x (B,C,F,T) fp32 -> xT[n][t][c] bf16, zero-padded rows ----
__global__ __launch_bounds__(256) void xpose_bf16(const float* __restrict__ x,
                                                  u16* __restrict__ xT) {
  const int n = blockIdx.x;     // 0..NPAD_-1
  const int t = threadIdx.x;    // 0..255
  __align__(16) u16 buf[C_];
  if (n < N_) {
    const int bb = n / F_, ff = n % F_;
    const float* src = x + (((size_t)bb * C_) * F_ + ff) * T_ + t;
    #pragma unroll 8
    for (int c = 0; c < C_; ++c)
      buf[c] = f2bf(src[(size_t)c * F_ * T_]);   // coalesced across t
  } else {
    #pragma unroll
    for (int c = 0; c < C_; ++c) buf[c] = 0;
  }
  u16* dst = xT + ((size_t)n * T_ + t) * C_;
  #pragma unroll
  for (int i = 0; i < 8; ++i)
    ((v4u*)dst)[i] = ((const v4u*)buf)[i];       // 128B per thread
}

// ---------------- Kernel 2: weights fp32 -> bf16, fold biases ------------------------
__global__ __launch_bounds__(256) void wconv(const float* __restrict__ Wih,
                                             const float* __restrict__ Whh,
                                             const float* __restrict__ Wlin,
                                             const float* __restrict__ bih,
                                             const float* __restrict__ bhh,
                                             u16* __restrict__ wih_o,
                                             u16* __restrict__ whh_o,
                                             u16* __restrict__ wlin_o,
                                             float* __restrict__ bcomb) {
  const int i = blockIdx.x * 256 + threadIdx.x;
  if (i < 4 * H_ * C_) wih_o[i]  = f2bf(Wih[i]);
  if (i < 4 * H_ * H_) whh_o[i]  = f2bf(Whh[i]);
  if (i < OUT_ * H_)   wlin_o[i] = f2bf(Wlin[i]);
  if (i < 4 * H_)      bcomb[i]  = bih[i] + bhh[i];
}

// ---------------- Kernel 3: persistent WMMA LSTM + fused projection ------------------
// One WG per 16-sequence tile; 8 waves; wave w owns gate columns [32w,32w+32) of each
// of the 4 gate blocks (i,f,g,o), so the cell update is elementwise in D layout.
// W_hh/W_ih stream from L2 every step (512KB resident in 192MB L2); laundered base
// pointers stop the compiler from hoisting+spilling the fragments.
__global__ __launch_bounds__(256) void lstm_persist(
    const u16* __restrict__ xT, const u16* __restrict__ Wih,
    const u16* __restrict__ Whh, const u16* __restrict__ Wlin,
    const float* __restrict__ bcomb, const float* __restrict__ blin,
    float* __restrict__ out) {
  __shared__ __align__(16) u16 h_lds[16 * H_];   // 8 KB: h_t tile, row-major [m][H]

  const int tid  = threadIdx.x;
  const int w    = tid >> 5;    // wave 0..7 (wave32)
  const int lane = tid & 31;
  const int r    = lane & 15;
  const int hi   = lane >> 4;
  const int n0   = blockIdx.x * 16;
  const int j0   = w * 32;      // h-column base owned by this wave

  for (int i = tid; i < 16 * H_; i += 256) h_lds[i] = 0;   // h_{-1} = 0
  __syncthreads();

  v8f cst[2];                                              // c state, per subtile
  #pragma unroll
  for (int s = 0; s < 2; ++s)
    #pragma unroll
    for (int k = 0; k < 8; ++k) cst[s][k] = 0.f;

  const u16* xrowbase = xT + (size_t)(n0 + r) * (T_ * C_);

  // Per-lane bias values (truly invariant; cheap to keep resident: 8 VGPRs + 1)
  float bias[4][2];
  #pragma unroll
  for (int gt = 0; gt < 4; ++gt)
    #pragma unroll
    for (int s = 0; s < 2; ++s)
      bias[gt][s] = bcomb[gt * 256 + j0 + s * 16 + r];
  const float blv = blin[(w & 3) * 16 + r];

  for (int t = 0; t < T_; ++t) {
    // Laundered (non-hoistable) global weight bases for this step
    const GAS u16* WihT = launder_g(Wih);
    const GAS u16* WhhT = launder_g(Whh);

    // ---- gates = bias + x_t W_ih^T + h_{t-1} W_hh^T  (16 x 128 per wave) ----
    v8f acc[4][2];
    #pragma unroll
    for (int gt = 0; gt < 4; ++gt)
      #pragma unroll
      for (int s = 0; s < 2; ++s) {
        #pragma unroll
        for (int k = 0; k < 8; ++k) acc[gt][s][k] = bias[gt][s];
      }

    const u16* xrow = xrowbase + t * C_;
    #pragma unroll
    for (int k0 = 0; k0 < C_; k0 += 32) {                  // K = C = 64: 2 steps
      v16bf a = load_frag_a(xrow + k0 + hi * 8);
      #pragma unroll
      for (int gt = 0; gt < 4; ++gt)
        #pragma unroll
        for (int s = 0; s < 2; ++s) {
          const GAS u16* brow =
              WihT + (size_t)(gt * 256 + j0 + s * 16 + r) * C_ + k0 + hi * 8;
          acc[gt][s] = wmma_bf16(a, load_frag_b_g(brow), acc[gt][s]);
        }
    }

    #pragma unroll
    for (int k0 = 0; k0 < H_; k0 += 32) {                  // K = H = 256: 8 steps
      v16bf a = load_frag_a(&h_lds[r * H_ + k0 + hi * 8]); // shared A across 8 tiles
      #pragma unroll
      for (int gt = 0; gt < 4; ++gt)
        #pragma unroll
        for (int s = 0; s < 2; ++s) {
          const GAS u16* brow =
              WhhT + (size_t)(gt * 256 + j0 + s * 16 + r) * H_ + k0 + hi * 8;
          acc[gt][s] = wmma_bf16(a, load_frag_b_g(brow), acc[gt][s]);  // L2 stream
        }
    }

    __syncthreads();   // all h_{t-1} fragment reads complete

    // ---- elementwise cell update in D layout; write h_t to LDS ----
    #pragma unroll
    for (int s = 0; s < 2; ++s) {
      const int j = j0 + s * 16 + r;
      #pragma unroll
      for (int k = 0; k < 8; ++k) {
        const float ig = sigm(acc[0][s][k]);
        const float fg = sigm(acc[1][s][k]);
        const float gg = tanh_fast(acc[2][s][k]);
        const float og = sigm(acc[3][s][k]);
        const float cc = fg * cst[s][k] + ig * gg;
        cst[s][k] = cc;
        const int m = k + 8 * hi;                          // D row for this vgpr/lane
        h_lds[m * H_ + j] = f2bf(og * tanh_fast(cc));
      }
    }

    __syncthreads();   // h_t visible

    // ---- fused projection: out_t = h_t W_lin^T + b_lin (waves 0..3, one o-tile each)
    if (w < 4) {
      const GAS u16* WlinT = launder_g(Wlin);
      const int ocol = w * 16 + r;
      v8f oacc;
      #pragma unroll
      for (int k = 0; k < 8; ++k) oacc[k] = blv;
      #pragma unroll
      for (int k0 = 0; k0 < H_; k0 += 32) {
        v16bf a = load_frag_a(&h_lds[r * H_ + k0 + hi * 8]);
        const GAS u16* brow = WlinT + (size_t)ocol * H_ + k0 + hi * 8;
        oacc = wmma_bf16(a, load_frag_b_g(brow), oacc);
      }
      #pragma unroll
      for (int k = 0; k < 8; ++k) {
        const int m = k + 8 * hi;
        const int n = n0 + m;
        if (n < N_) {
          const int bb = n / F_, ff = n % F_;
          out[(((size_t)bb * OUT_ + ocol) * F_ + ff) * T_ + t] = oacc[k];
        }
      }
    }
  }
}

extern "C" void kernel_launch(void* const* d_in, const int* in_sizes, int n_in,
                              void* d_out, int out_size, void* d_ws, size_t ws_size,
                              hipStream_t stream) {
  (void)in_sizes; (void)n_in; (void)out_size; (void)ws_size;
  const float* x    = (const float*)d_in[0];
  const float* Wih  = (const float*)d_in[1];
  const float* Whh  = (const float*)d_in[2];
  const float* bih  = (const float*)d_in[3];
  const float* bhh  = (const float*)d_in[4];
  const float* Wlin = (const float*)d_in[5];
  const float* blin = (const float*)d_in[6];
  float* out = (float*)d_out;

  char* ws = (char*)d_ws;
  const size_t SZ_XT   = (size_t)NPAD_ * T_ * C_ * sizeof(u16);   // 34,078,720
  const size_t OFF_WIH = SZ_XT;
  const size_t OFF_WHH = OFF_WIH + (size_t)4 * H_ * C_ * sizeof(u16);
  const size_t OFF_WLN = OFF_WHH + (size_t)4 * H_ * H_ * sizeof(u16);
  const size_t OFF_BC  = OFF_WLN + (size_t)OUT_ * H_ * sizeof(u16);

  u16*   xT     = (u16*)(ws);
  u16*   wih_b  = (u16*)(ws + OFF_WIH);
  u16*   whh_b  = (u16*)(ws + OFF_WHH);
  u16*   wlin_b = (u16*)(ws + OFF_WLN);
  float* bcomb  = (float*)(ws + OFF_BC);

  xpose_bf16<<<NPAD_, 256, 0, stream>>>(x, xT);
  wconv<<<(4 * H_ * H_) / 256, 256, 0, stream>>>(Wih, Whh, Wlin, bih, bhh,
                                                 wih_b, whh_b, wlin_b, bcomb);
  lstm_persist<<<NBLK_, 256, 0, stream>>>(xT, wih_b, whh_b, wlin_b, bcomb, blin, out);
}
